// SelfAttention_6519760355658
// MI455X (gfx1250) — compile-verified, matrix operand order checked by reading, not run on hardware
//
#include <hip/hip_runtime.h>
#include <math.h>

typedef __bf16 bf16_t;
typedef __attribute__((ext_vector_type(8)))  __bf16 v8bf;
typedef __attribute__((ext_vector_type(16))) __bf16 v16bf;
typedef __attribute__((ext_vector_type(8)))  float  v8f;
typedef __attribute__((ext_vector_type(4)))  unsigned int u32x4;
typedef __attribute__((ext_vector_type(8)))  int i32x8;
typedef __attribute__((ext_vector_type(4)))  int i32x4;

#define BB 4
#define CC 256
#define NN 4096
#define OO 768   // 3*C

// Load a 16-bit A/B WMMA fragment (K=32) for this lane.
// p must already point at: base + (row_or_col)*ld + c0 + (lane>>4)*8
// Per the CDNA5 ISA 16-bit layout, each lane holds K-chunk [hi*8, hi*8+8)
// in halves 0..7 and K-chunk [16+hi*8, 16+hi*8+8) in halves 8..15.
__device__ __forceinline__ v16bf load_frag(const bf16_t* p) {
    v8bf lo = *(const v8bf*)p;
    v8bf hi = *(const v8bf*)(p + 16);
    return __builtin_shufflevector(lo, hi, 0,1,2,3,4,5,6,7,8,9,10,11,12,13,14,15);
}

__device__ __forceinline__ v8f wmma_bf16(v16bf a, v16bf b, v8f c) {
    return __builtin_amdgcn_wmma_f32_16x16x32_bf16(false, a, false, b, (short)0, c,
                                                   false, false);
}

// Async copy of 16 bytes global -> LDS (per-lane addresses), tracked by ASYNCcnt.
__device__ __forceinline__ void async_ld_b128(unsigned lds_off, const void* gsrc) {
    asm volatile("global_load_async_to_lds_b128 %0, %1, off"
                 :: "v"(lds_off), "v"(gsrc) : "memory");
}

__device__ __forceinline__ void wait_asynccnt0() {
#if __has_builtin(__builtin_amdgcn_s_wait_asynccnt)
    __builtin_amdgcn_s_wait_asynccnt(0);
#else
    asm volatile("s_wait_asynccnt 0x0" ::: "memory");
#endif
}

// TDM: load a 2D bf16 tile (tile_d0 elems x tile_d1 rows, row stride = srcStride
// elements) from global to LDS.  D# per CDNA5 ISA ch.8.  Wave-uniform args.
// This toolchain's builtin is the 6-arg form:
//   (u32x4 g0, i32x8 g1, i32x4 g2, i32x4 g3, i32x8 pad, i32 cpol)
__device__ __forceinline__ void tdm_load_tile_bf16(unsigned lds_addr,
                                                   const bf16_t* gsrc,
                                                   unsigned tile_d0, unsigned tile_d1,
                                                   unsigned srcStride) {
    unsigned long long ga = (unsigned long long)(size_t)gsrc;
    u32x4 g0;
    g0[0] = 1u;                                   // count=1, no gather
    g0[1] = lds_addr;                             // LDS byte address
    g0[2] = (unsigned)ga;                         // global addr [31:0]
    g0[3] = ((unsigned)(ga >> 32) & 0x01FFFFFFu)  // global addr [56:32]
            | (2u << 30);                         // type = 2 ("image")
    i32x8 g1;
    g1[0] = (int)(1u << 16);                      // data_size=1 (2 bytes), mask=0
    g1[1] = (int)((CC & 0xFFFFu) << 16);          // tensor_dim0[15:0] (=256)
    g1[2] = (int)((CC >> 16) | ((NN & 0xFFFFu) << 16));   // dim0 hi | dim1 lo
    g1[3] = (int)((NN >> 16) | (tile_d0 << 16));  // dim1 hi | tile_dim0
    g1[4] = (int)(tile_d1);                       // tile_dim1 | tile_dim2=0
    g1[5] = (int)srcStride;                       // tensor_dim0_stride[31:0]
    g1[6] = 0;                                    // stride0 hi | stride1 lo
    g1[7] = 0;                                    // stride1 hi
#if __has_builtin(__builtin_amdgcn_tensor_load_to_lds)
    i32x4 z4 = {0, 0, 0, 0};
    i32x8 z8 = {0, 0, 0, 0, 0, 0, 0, 0};
    __builtin_amdgcn_tensor_load_to_lds(g0, g1, z4, z4, z8, 0);
#else
    asm volatile("tensor_load_to_lds %0, %1" :: "s"(g0), "s"(g1) : "memory");
#endif
}

// ---------------- fp32 -> bf16 conversions ----------------

// x (b,c,n) fp32 -> xbT (b,n,c) bf16  (transpose so GEMM B-operand is K-contiguous)
__global__ void cvt_x(const float* __restrict__ x, bf16_t* __restrict__ xbT) {
    size_t idx = (size_t)blockIdx.x * 256 + threadIdx.x;   // B*C*N total
    int n = (int)(idx % NN);
    int c = (int)((idx / NN) % CC);
    int b = (int)(idx / ((size_t)NN * CC));
    xbT[((size_t)(b * NN + n)) * CC + c] = (bf16_t)x[idx];
}

__global__ void cvt_w(const float* __restrict__ wq, const float* __restrict__ wp,
                      bf16_t* __restrict__ wqb, bf16_t* __restrict__ wpb) {
    int idx = blockIdx.x * 256 + threadIdx.x;
    if (idx < OO * CC) wqb[idx] = (bf16_t)wq[idx];
    int j = idx - OO * CC;
    if (j >= 0 && j < CC * CC) wpb[j] = (bf16_t)wp[j];
}

// ---------------- QKV projection ----------------
// qkv[o][n] = sum_c wq[o][c] * x[c][n] + b_qkv[o]
// Writes: Qt (b,n,c) bf16, Kt (b,n,c) bf16, V (b,c,n) bf16
__global__ void qkv_gemm(const bf16_t* __restrict__ xbT, const bf16_t* __restrict__ wq,
                         const float* __restrict__ b_qkv,
                         bf16_t* __restrict__ Qt, bf16_t* __restrict__ Kt,
                         bf16_t* __restrict__ V) {
    const int l   = threadIdx.x;
    const int col = l & 15;
    const int hi  = l >> 4;
    const int n0  = blockIdx.x * 16;
    const int o0  = blockIdx.y * 16;
    const int b   = blockIdx.z;

    const bf16_t* arow = wq  + (size_t)(o0 + col) * CC + hi * 8;
    const bf16_t* brow = xbT + ((size_t)(b * NN + n0 + col)) * CC + hi * 8;

    v8f acc = {};
#pragma unroll
    for (int c0 = 0; c0 < CC; c0 += 32) {
        v16bf a  = load_frag(arow + c0);
        v16bf bm = load_frag(brow + c0);
        acc = wmma_bf16(a, bm, acc);
    }

    const int n     = n0 + col;
    const int obase = o0 + hi * 8;     // D row = r + hi*8
    if (o0 < CC) {                     // ---- Q, transposed (n,c)
        bf16_t* qp = Qt + ((size_t)(b * NN + n)) * CC + obase;
#pragma unroll
        for (int r = 0; r < 8; ++r) qp[r] = (bf16_t)(acc[r] + b_qkv[obase + r]);
    } else if (o0 < 2 * CC) {          // ---- K, transposed (n,c)
        bf16_t* kp = Kt + ((size_t)(b * NN + n)) * CC + (obase - CC);
#pragma unroll
        for (int r = 0; r < 8; ++r) kp[r] = (bf16_t)(acc[r] + b_qkv[obase + r]);
    } else {                           // ---- V, (c,n)
#pragma unroll
        for (int r = 0; r < 8; ++r) {
            int o = obase - 2 * CC + r;
            V[((size_t)(b * CC + o)) * NN + n] = (bf16_t)(acc[r] + b_qkv[obase + r]);
        }
    }
}

// ---------------- S = (Q^T K) / sqrt(C) ----------------
// Block = 256 threads (8 waves) computing a 64(i) x 32(j) S region.
// Per k-step: Q tile (64x32) staged via async-to-LDS, K tile (32x32) via TDM.
__global__ void sqk_gemm(const bf16_t* __restrict__ Qt, const bf16_t* __restrict__ Kt,
                         float* __restrict__ S, int b) {
    __shared__ bf16_t qs[64 * 32];   // 4 KB
    __shared__ bf16_t ks[32 * 32];   // 2 KB

    const int t   = threadIdx.x;     // 0..255
    const int l   = t & 31;
    const int w   = t >> 5;          // wave 0..7
    const int col = l & 15;
    const int hi  = l >> 4;
    const int j0  = blockIdx.x * 32;
    const int i0  = blockIdx.y * 64;
    const int i_sub = (w & 3) * 16;
    const int j_sub = (w >> 2) * 16;

    const unsigned qs_base = (unsigned)(size_t)(&qs[0]);
    const unsigned ks_base = (unsigned)(size_t)(&ks[0]);

    // per-thread Q chunk: 64 rows x 4 x 16B segments
    const int qrow = t >> 2;         // 0..63
    const int qseg = t & 3;          // 0..3
    const bf16_t* qsrc = Qt + ((size_t)(b * NN + i0 + qrow)) * CC + qseg * 8;
    const unsigned q_lds = qs_base + t * 16;

    v8f acc = {};
    for (int c0 = 0; c0 < CC; c0 += 32) {
        __syncthreads();                       // previous iteration's reads done
        if (t < 32) {                          // wave 0: K tile 32x32 via TDM
            tdm_load_tile_bf16(ks_base, Kt + ((size_t)(b * NN + j0)) * CC + c0,
                               32u, 32u, (unsigned)CC);
        }
        async_ld_b128(q_lds, qsrc + c0);       // all waves: Q tile via async copy
        wait_asynccnt0();
        if (t < 32) __builtin_amdgcn_s_wait_tensorcnt(0);
        __syncthreads();                       // tiles visible to all waves

        v16bf a  = load_frag(qs + (i_sub + col) * 32 + hi * 8);  // ds_load
        v16bf bm = load_frag(ks + (j_sub + col) * 32 + hi * 8);  // ds_load
        acc = wmma_bf16(a, bm, acc);
    }

    const float scale = 0.0625f;   // 1/sqrt(256)
#pragma unroll
    for (int r = 0; r < 8; ++r)
        S[(size_t)(i0 + i_sub + hi * 8 + r) * NN + j0 + j_sub + col] = acc[r] * scale;
}

// ---------------- row softmax (fp32, in place) ----------------
__global__ void softmax_rows(float* __restrict__ S) {
    __shared__ float red[256];
    const int row = blockIdx.x;
    const int t   = threadIdx.x;
    float* p = S + (size_t)row * NN;

    float m = -1e30f;
    for (int j = t; j < NN; j += 256) m = fmaxf(m, p[j]);
    red[t] = m; __syncthreads();
    for (int s = 128; s > 0; s >>= 1) {
        if (t < s) red[t] = fmaxf(red[t], red[t + s]);
        __syncthreads();
    }
    m = red[0]; __syncthreads();

    float sum = 0.f;
    for (int j = t; j < NN; j += 256) {
        float e = __expf(p[j] - m);
        p[j] = e;
        sum += e;
    }
    red[t] = sum; __syncthreads();
    for (int s = 128; s > 0; s >>= 1) {
        if (t < s) red[t] += red[t + s];
        __syncthreads();
    }
    const float inv = 1.0f / red[0];
    for (int j = t; j < NN; j += 256) p[j] *= inv;
}

// ---------------- O = P * V^T ----------------
// Ht[n][c] = sum_j P[i=n][j] * V[c][j]   (one batch; P read fp32, cvt to bf16)
__global__ void pv_gemm(const float* __restrict__ S, const bf16_t* __restrict__ V,
                        bf16_t* __restrict__ Ht, int b) {
    const int l   = threadIdx.x;
    const int col = l & 15;
    const int hi  = l >> 4;
    const int c0t = blockIdx.x * 16;
    const int i0  = blockIdx.y * 16;

    const float*  srow = S + (size_t)(i0 + col) * NN + hi * 8;
    const bf16_t* vrow = V + ((size_t)(b * CC + c0t + col)) * NN + hi * 8;

    v8f acc = {};
    for (int j0 = 0; j0 < NN; j0 += 32) {
        __builtin_prefetch(vrow + j0 + 64, 0, 1);   // global_prefetch_b8
        v16bf a;
#pragma unroll
        for (int t2 = 0; t2 < 8; ++t2) {
            a[t2]     = (bf16_t)srow[j0 + t2];
            a[t2 + 8] = (bf16_t)srow[j0 + 16 + t2];
        }
        v16bf bm = load_frag(vrow + j0);
        acc = wmma_bf16(a, bm, acc);
    }

#pragma unroll
    for (int r = 0; r < 8; ++r)
        Ht[((size_t)(b * NN + i0 + hi * 8 + r)) * CC + c0t + col] = (bf16_t)acc[r];
}

// ---------------- output projection + residual ----------------
// out[b][o][n] = sum_c wp[o][c] * H[c][n] + b_proj[o] + x[b][o][n]
__global__ void proj_gemm(const bf16_t* __restrict__ wp, const bf16_t* __restrict__ Ht,
                          const float* __restrict__ b_proj, const float* __restrict__ x,
                          float* __restrict__ out) {
    const int l   = threadIdx.x;
    const int col = l & 15;
    const int hi  = l >> 4;
    const int n0  = blockIdx.x * 16;
    const int o0  = blockIdx.y * 16;
    const int b   = blockIdx.z;

    const bf16_t* arow = wp + (size_t)(o0 + col) * CC + hi * 8;
    const bf16_t* brow = Ht + ((size_t)(b * NN + n0 + col)) * CC + hi * 8;

    v8f acc = {};
#pragma unroll
    for (int c0 = 0; c0 < CC; c0 += 32) {
        v16bf a  = load_frag(arow + c0);
        v16bf bm = load_frag(brow + c0);
        acc = wmma_bf16(a, bm, acc);
    }

#pragma unroll
    for (int r = 0; r < 8; ++r) {
        int o = o0 + hi * 8 + r;
        size_t idx = ((size_t)(b * CC + o)) * NN + n0 + col;
        out[idx] = acc[r] + b_proj[o] + x[idx];
    }
}

// ---------------- host launch ----------------
extern "C" void kernel_launch(void* const* d_in, const int* in_sizes, int n_in,
                              void* d_out, int out_size, void* d_ws, size_t ws_size,
                              hipStream_t stream) {
    const float* x      = (const float*)d_in[0];
    const float* w_qkv  = (const float*)d_in[1];
    const float* b_qkv  = (const float*)d_in[2];
    const float* w_proj = (const float*)d_in[3];
    const float* b_proj = (const float*)d_in[4];
    float* out = (float*)d_out;

    char* ws = (char*)d_ws;
    bf16_t* xbT = (bf16_t*)(ws);                 //  8,388,608 B
    bf16_t* wqb = (bf16_t*)(ws +  8388608);      //    393,216 B
    bf16_t* wpb = (bf16_t*)(ws +  8781824);      //    131,072 B
    bf16_t* Qt  = (bf16_t*)(ws +  8912896);      //  8,388,608 B
    bf16_t* Kt  = (bf16_t*)(ws + 17301504);      //  8,388,608 B
    bf16_t* V   = (bf16_t*)(ws + 25690112);      //  8,388,608 B
    bf16_t* Ht  = (bf16_t*)(ws + 34078720);      //  8,388,608 B
    float*  S   = (float*) (ws + 42467328);      // 67,108,864 B (per-batch, reused)

    cvt_x<<<(BB * CC * NN) / 256, 256, 0, stream>>>(x, xbT);
    cvt_w<<<(OO * CC + CC * CC + 255) / 256, 256, 0, stream>>>(w_qkv, w_proj, wqb, wpb);

    qkv_gemm<<<dim3(NN / 16, OO / 16, BB), 32, 0, stream>>>(xbT, wqb, b_qkv, Qt, Kt, V);

    for (int b = 0; b < BB; ++b) {   // stream order serializes; S buffer reused
        sqk_gemm<<<dim3(NN / 32, NN / 64), 256, 0, stream>>>(Qt, Kt, S, b);
        softmax_rows<<<NN, 256, 0, stream>>>(S);
        pv_gemm<<<dim3(CC / 16, NN / 16), 32, 0, stream>>>(S, V, Ht, b);
    }

    proj_gemm<<<dim3(NN / 16, CC / 16, BB), 32, 0, stream>>>(wpb, Ht, b_proj, x, out);
}
